// GNNActorOld_3676492005377
// MI455X (gfx1250) — matrix operand order repaired and checked by reading, not compiled, not run on hardware
//
#include <hip/hip_runtime.h>
#include <hip/hip_bf16.h>

typedef __attribute__((ext_vector_type(16))) _Float16 v16h;
typedef __attribute__((ext_vector_type(8)))  _Float16 v8h;
typedef __attribute__((ext_vector_type(8)))  float    v8f;
typedef __attribute__((ext_vector_type(4)))  float    v4f;

#define CDIM 128
#define HDIM 32
#define ACTD 6

#define WSTRIDE (CDIM + 8)   // 136 halves = 272 B rows: 16B-aligned, bank-rotating
#define W1S     (CDIM + 8)   // w1t row stride
#define W2S     (HDIM + 8)   // 40 halves = 80 B: 16B-aligned, bank-rotating
#define Y1S     (HDIM + 8)
#define Y2S     (HDIM + 4)   // 36 floats = 144 B: 16B-aligned, bank-rotating

__device__ __forceinline__ v8f v8f_zero() {
  v8f z = {0.f, 0.f, 0.f, 0.f, 0.f, 0.f, 0.f, 0.f};
  return z;
}

__device__ __forceinline__ v16h concat8(v8h lo, v8h hi) {
  return __builtin_shufflevector(lo, hi, 0, 1, 2, 3, 4, 5, 6, 7,
                                 8, 9, 10, 11, 12, 13, 14, 15);
}

// ---------------------------------------------------------------- utilities
__global__ __launch_bounds__(256) void k_zero(float* __restrict__ p, int n) {
  int i = blockIdx.x * 256 + threadIdx.x;
  int stride = gridDim.x * 256;
  for (; i < n; i += stride) p[i] = 0.f;
}

__global__ __launch_bounds__(256) void k_degree(const int* __restrict__ dst,
                                                float* __restrict__ deg, int E) {
  int e = blockIdx.x * 256 + threadIdx.x;
  if (e < E) atomicAdd(&deg[dst[e]], 1.0f);
}

// deg -> dinv = rsqrt(deg + 1)  (the +1 accounts for the self loop)
__global__ __launch_bounds__(256) void k_dinv(float* __restrict__ deg, int n) {
  int i = blockIdx.x * 256 + threadIdx.x;
  if (i < n) deg[i] = rsqrtf(deg[i] + 1.0f);
}

__global__ __launch_bounds__(256) void k_cvt16(const float* __restrict__ x,
                                               _Float16* __restrict__ y, int n) {
  int i = blockIdx.x * 256 + threadIdx.x;
  if (i < n) y[i] = (_Float16)x[i];
}

// ---------------------------------------------------------------- h = x @ W_gcn
// 256 threads = 8 waves; block covers 128 rows x 128 cols; K=128 in 4 WMMA steps.
// W staged transposed (Wt[n][k]) so each B fragment is 2x ds_load_b128; all 8
// B fragments of a k-step are loaded before the WMMA burst to batch dscnt waits.
__global__ __launch_bounds__(256) void k_gemm_gcn(const _Float16* __restrict__ xh,
                                                  const float* __restrict__ Wg,
                                                  float* __restrict__ h, int n_rows) {
  __shared__ __align__(16) _Float16 Wt[CDIM * WSTRIDE];
  for (int i = threadIdx.x; i < CDIM * CDIM; i += 256) {
    int k = i >> 7, n = i & 127;                 // Wg row-major [k][n], coalesced read
    Wt[n * WSTRIDE + k] = (_Float16)Wg[i];
  }
  __syncthreads();

  const int wave  = threadIdx.x >> 5;
  const int lane  = threadIdx.x & 31;
  const int half  = lane >> 4;   // 0 or 1
  const int lan16 = lane & 15;

  int row  = blockIdx.x * 128 + wave * 16 + lan16;
  int rowc = row < n_rows ? row : n_rows - 1;    // clamp: EXEC must stay all-ones
  const int khi = half * 8;
  const _Float16* xrow = xh + (size_t)rowc * CDIM;
  const _Float16* wbase = Wt + lan16 * WSTRIDE + half * 8;

  v8f acc[8];
#pragma unroll
  for (int t = 0; t < 8; ++t) acc[t] = v8f_zero();

#pragma unroll
  for (int ks = 0; ks < 4; ++ks) {
    const int kb = ks * 32;
    // A fragment 16x32 f16: lane holds row lan16, two contiguous 8-half runs
    const _Float16* xr = xrow + kb + khi;
    v16h a = concat8(*(const v8h*)xr, *(const v8h*)(xr + 16));
    // Load all 8 B fragments first (16x ds_load_b128), then WMMA burst
    v16h b[8];
#pragma unroll
    for (int t = 0; t < 8; ++t) {
      const _Float16* wp = wbase + t * (16 * WSTRIDE) + kb;
      b[t] = concat8(*(const v8h*)wp, *(const v8h*)(wp + 16));
    }
#pragma unroll
    for (int t = 0; t < 8; ++t)
      acc[t] = __builtin_amdgcn_wmma_f32_16x16x32_f16(false, a, false, b[t],
                                                      (short)0, acc[t], false, false);
  }

  // C/D layout: vgpr r -> row r + 8*half, col lan16
  const int rbase = blockIdx.x * 128 + wave * 16 + half * 8;
#pragma unroll
  for (int t = 0; t < 8; ++t)
#pragma unroll
    for (int r = 0; r < 8; ++r) {
      int gr = rbase + r;
      if (gr < n_rows) h[(size_t)gr * CDIM + t * 16 + lan16] = acc[t][r];
    }
}

// ---------------------------------------------------------------- scatter-add
// one wave per edge, 4 coalesced float atomics per lane (128 channels)
__global__ __launch_bounds__(256) void k_scatter(const int* __restrict__ src,
                                                 const int* __restrict__ dst,
                                                 const float* __restrict__ dinv,
                                                 const float* __restrict__ h,
                                                 float* __restrict__ agg, int E) {
  int eg = blockIdx.x * 8 + (threadIdx.x >> 5);
  if (eg >= E) return;
  int lane = threadIdx.x & 31;
  int s = src[eg], d = dst[eg];
  float norm = dinv[s] * dinv[d];
  const float* hs = h + (size_t)s * CDIM;
  float* ad = agg + (size_t)d * CDIM;
#pragma unroll
  for (int j = 0; j < 4; ++j) {
    int c = lane + 32 * j;
    atomicAdd(&ad[c], norm * hs[c]);
  }
}

// ---------------------------------------------------------------- fused actor MLP
// x_res = relu(agg + dinv^2*h + b_gcn) + state ; two WMMA layers; dot+softplus
__global__ __launch_bounds__(256) void k_mlp(
    const float* __restrict__ state, const float* __restrict__ h,
    const float* __restrict__ agg, const float* __restrict__ dinv,
    const float* __restrict__ b_gcn,
    const float* __restrict__ w1, const float* __restrict__ b1,
    const float* __restrict__ w2, const float* __restrict__ b2,
    const float* __restrict__ w3, const float* __restrict__ b3,
    float* __restrict__ conc, int n_rows) {
  __shared__ __align__(16) _Float16 w1t[HDIM * W1S];     // transposed [n][k]
  __shared__ __align__(16) _Float16 w2t[HDIM * W2S];     // transposed [n][k]
  __shared__ __align__(16) _Float16 y1t[8][16 * Y1S];    // per-wave y1, row-major [m][k]
  __shared__ __align__(16) float    y2s[8][16 * Y2S];    // per-wave y2 for final dot
  for (int i = threadIdx.x; i < CDIM * HDIM; i += 256) {
    int k = i >> 5, n = i & 31;                          // w1 row-major [k][n]
    w1t[n * W1S + k] = (_Float16)w1[i];
  }
  for (int i = threadIdx.x; i < HDIM * HDIM; i += 256) {
    int k = i >> 5, n = i & 31;
    w2t[n * W2S + k] = (_Float16)w2[i];
  }
  __syncthreads();

  const int wave  = threadIdx.x >> 5;
  const int lane  = threadIdx.x & 31;
  const int half  = lane >> 4;
  const int lan16 = lane & 15;

  int row  = blockIdx.x * 128 + wave * 16 + lan16;
  int rowc = row < n_rows ? row : n_rows - 1;
  float di  = dinv[rowc];
  float dsq = di * di;  // self-loop norm = dinv[i]^2
  const int khi = half * 8;
  const size_t rbase = (size_t)rowc * CDIM;
  const float* pa = agg + rbase;
  const float* ph = h + rbase;
  const float* ps = state + rbase;

  // ---- layer 1: y1 = leaky_relu(x_res @ w1 + b1), K=128, N=32 (2 tiles)
  v8f acc1[2] = { v8f_zero(), v8f_zero() };
#pragma unroll
  for (int ks = 0; ks < 4; ++ks) {
    const int kb = ks * 32;
    v16h a;
#pragma unroll
    for (int q = 0; q < 2; ++q) {                        // two 8-wide runs: +0, +16
      const int c = kb + khi + q * 16;
      v4f a0 = *(const v4f*)(pa + c);
      v4f a1 = *(const v4f*)(pa + c + 4);
      v4f h0 = *(const v4f*)(ph + c);
      v4f h1 = *(const v4f*)(ph + c + 4);
      v4f s0 = *(const v4f*)(ps + c);
      v4f s1 = *(const v4f*)(ps + c + 4);
#pragma unroll
      for (int j = 0; j < 4; ++j) {
        float v = a0[j] + dsq * h0[j] + b_gcn[c + j];
        v = fmaxf(v, 0.f) + s0[j];                       // relu + residual
        a[q * 8 + j] = (_Float16)v;
        float w = a1[j] + dsq * h1[j] + b_gcn[c + 4 + j];
        w = fmaxf(w, 0.f) + s1[j];
        a[q * 8 + 4 + j] = (_Float16)w;
      }
    }
    const int koff = kb + half * 8;
    v16h b[2];
#pragma unroll
    for (int t = 0; t < 2; ++t) {
      const _Float16* wp = w1t + (t * 16 + lan16) * W1S + koff;
      b[t] = concat8(*(const v8h*)wp, *(const v8h*)(wp + 16));
    }
#pragma unroll
    for (int t = 0; t < 2; ++t)
      acc1[t] = __builtin_amdgcn_wmma_f32_16x16x32_f16(false, a, false, b[t],
                                                       (short)0, acc1[t], false, false);
  }
  // bias + leaky relu, stage into LDS row-major f16 (C-layout -> A-layout transpose)
#pragma unroll
  for (int t = 0; t < 2; ++t) {
    float bv = b1[t * 16 + lan16];
#pragma unroll
    for (int r = 0; r < 8; ++r) {
      float x = acc1[t][r] + bv;
      x = x > 0.f ? x : 0.01f * x;
      int m = r + half * 8;
      y1t[wave][m * Y1S + t * 16 + lan16] = (_Float16)x;
    }
  }
  __syncthreads();

  // ---- layer 2: y2 = leaky_relu(y1 @ w2 + b2), K=32 (single WMMA step)
  const _Float16* yp = &y1t[wave][lan16 * Y1S + khi];
  v16h a2 = concat8(*(const v8h*)yp, *(const v8h*)(yp + 16));
  v8f acc2[2] = { v8f_zero(), v8f_zero() };
  {
    const int koff = half * 8;
    v16h b[2];
#pragma unroll
    for (int t = 0; t < 2; ++t) {
      const _Float16* wp = w2t + (t * 16 + lan16) * W2S + koff;
      b[t] = concat8(*(const v8h*)wp, *(const v8h*)(wp + 16));
    }
#pragma unroll
    for (int t = 0; t < 2; ++t)
      acc2[t] = __builtin_amdgcn_wmma_f32_16x16x32_f16(false, a2, false, b[t],
                                                       (short)0, acc2[t], false, false);
  }
#pragma unroll
  for (int t = 0; t < 2; ++t) {
    float bv = b2[t * 16 + lan16];
#pragma unroll
    for (int r = 0; r < 8; ++r) {
      float x = acc2[t][r] + bv;
      x = x > 0.f ? x : 0.01f * x;
      int m = r + half * 8;
      y2s[wave][m * Y2S + t * 16 + lan16] = x;
    }
  }
  __syncthreads();

  // ---- layer 3: conc = softplus(y2 @ w3 + b3) — 32->1 dot per row
  if (lane < 16) {
    int gr = blockIdx.x * 128 + wave * 16 + lane;
    if (gr < n_rows) {
      float s = b3[0];
#pragma unroll
      for (int n = 0; n < HDIM; n += 4) {
        v4f y = *(const v4f*)&y2s[wave][lane * Y2S + n];
        v4f w = *(const v4f*)(w3 + n);
        s += y[0] * w[0] + y[1] * w[1] + y[2] * w[2] + y[3] * w[3];
      }
      float sp = (s > 20.f) ? s : log1pf(expf(s));
      conc[gr] = sp;
    }
  }
}

// ---------------------------------------------------------------- normalize groups
__global__ __launch_bounds__(256) void k_action(const float* __restrict__ conc,
                                                float* __restrict__ out,
                                                float* __restrict__ sumbuf, int ngroups) {
  int g = blockIdx.x * 256 + threadIdx.x;
  if (g >= ngroups) return;
  float v[ACTD];
  float s = 0.f, sa = 0.f;
#pragma unroll
  for (int j = 0; j < ACTD; ++j) {
    v[j] = conc[g * ACTD + j];
    s += v[j];
    sa += fabsf(v[j]);
  }
  float inv = 1.0f / (s + 1e-20f);
#pragma unroll
  for (int j = 0; j < ACTD; ++j) out[g * ACTD + j] = v[j] * inv;
  atomicAdd(sumbuf, sa);
}

__global__ void k_final(const float* __restrict__ sumbuf,
                        float* __restrict__ out_reg, int n) {
  out_reg[0] = sumbuf[0] / (float)n;
}

// ---------------------------------------------------------------- launch
extern "C" void kernel_launch(void* const* d_in, const int* in_sizes, int n_in,
                              void* d_out, int out_size, void* d_ws, size_t ws_size,
                              hipStream_t stream) {
  const float* state = (const float*)d_in[0];
  const int*   eidx  = (const int*)d_in[1];
  const float* Wg    = (const float*)d_in[2];
  const float* bg    = (const float*)d_in[3];
  const float* w1    = (const float*)d_in[4];
  const float* b1    = (const float*)d_in[5];
  const float* w2    = (const float*)d_in[6];
  const float* b2    = (const float*)d_in[7];
  const float* w3    = (const float*)d_in[8];
  const float* b3    = (const float*)d_in[9];
  float* out = (float*)d_out;

  const int N  = in_sizes[0] / CDIM;
  const int E  = in_sizes[1] / 2;
  const int NG = N / ACTD;
  const int nc = N * CDIM;
  (void)n_in; (void)out_size; (void)ws_size;

  char* ws = (char*)d_ws;
  size_t off = 0;
  auto carve = [&](size_t bytes) {
    char* p = ws + off;
    off = (off + bytes + 255) & ~(size_t)255;
    return p;
  };
  float*    deg  = (float*)carve((size_t)N * 4);          // becomes dinv
  float*    h    = (float*)carve((size_t)nc * 4);         // x @ W_gcn
  float*    agg  = (float*)carve((size_t)nc * 4);         // scatter accumulator
  _Float16* xh   = (_Float16*)carve((size_t)nc * 2);      // f16 state
  float*    conc = (float*)carve((size_t)N * 4);
  float*    sum  = (float*)carve(256);

  const int* src = eidx;
  const int* dst = eidx + E;

  k_zero<<<dim3(1024), dim3(256), 0, stream>>>(deg, N);
  k_zero<<<dim3(4096), dim3(256), 0, stream>>>(agg, nc);
  k_zero<<<dim3(1), dim3(256), 0, stream>>>(sum, 64);
  k_degree<<<dim3((E + 255) / 256), dim3(256), 0, stream>>>(dst, deg, E);
  k_dinv<<<dim3((N + 255) / 256), dim3(256), 0, stream>>>(deg, N);
  k_cvt16<<<dim3((nc + 255) / 256), dim3(256), 0, stream>>>(state, xh, nc);
  k_gemm_gcn<<<dim3((N + 127) / 128), dim3(256), 0, stream>>>(xh, Wg, h, N);
  k_scatter<<<dim3((E + 7) / 8), dim3(256), 0, stream>>>(src, dst, deg, h, agg, E);
  k_mlp<<<dim3((N + 127) / 128), dim3(256), 0, stream>>>(state, h, agg, deg, bg,
                                                         w1, b1, w2, b2, w3, b3,
                                                         conc, N);
  k_action<<<dim3((NG + 255) / 256), dim3(256), 0, stream>>>(conc, out, sum, NG);
  k_final<<<dim3(1), dim3(1), 0, stream>>>(sum, out + (size_t)NG * ACTD, N);
}
